// MultiHeadSelfAttention_19250043420856
// MI455X (gfx1250) — matrix operand order repaired and checked by reading, not compiled, not run on hardware
//
#include <hip/hip_runtime.h>
#include <hip/hip_bf16.h>

typedef __attribute__((ext_vector_type(16))) __bf16 v16bf;
typedef __attribute__((ext_vector_type(8)))  __bf16 v8bf;
typedef __attribute__((ext_vector_type(8)))  float  v8f;

#define WMMA_BF16(a, b, c) \
  __builtin_amdgcn_wmma_f32_16x16x32_bf16(false, (a), false, (b), (short)0, (c), false, false)

static constexpr int BATCH = 4, SEQ = 2048, EMB = 768, HEADS = 12, HD = 64;
static constexpr int M_ROWS = BATCH * SEQ;   // 8192
static constexpr int QKV_OUT = 3 * EMB;      // 2304

// ---------- fragment loaders (per CDNA5 16-bit WMMA VGPR layouts) ----------

__device__ __forceinline__ v16bf cat8(v8bf lo, v8bf hi) {
  return __builtin_shufflevector(lo, hi, 0,1,2,3,4,5,6,7,8,9,10,11,12,13,14,15);
}

// A-matrix 16x32 (MxK), row-major source with leading dim ld.
// lanes 0-15: row=lane, K = 0..7 then 16..23 ; lanes 16-31: row=lane-16, K = 8..15 then 24..31
__device__ __forceinline__ v16bf load_a_frag(const __bf16* base, int ld, int lane) {
  const int row = lane & 15;
  const int kb  = (lane >> 4) << 3;           // 0 or 8
  const __bf16* p = base + (size_t)row * ld + kb;
  return cat8(*(const v8bf*)p, *(const v8bf*)(p + 16));
}

// B-matrix 32x16 (KxN) where memory is laid out [n][k] row-major (B[k][n] = M[n][k]).
// lanes 0-15: col=lane, K = 0..7 then 16..23 ; lanes 16-31: col=lane-16, K = 8..15 then 24..31
__device__ __forceinline__ v16bf load_b_frag(const __bf16* base, int ld, int lane) {
  const int col = lane & 15;
  const int kb  = (lane >> 4) << 3;           // 0 or 8
  const __bf16* p = base + (size_t)col * ld + kb;
  return cat8(*(const v8bf*)p, *(const v8bf*)(p + 16));
}

// ---------- kernel 0: fp32 -> bf16 convert ----------

__global__ void f32_to_bf16(const float* __restrict__ in, __bf16* __restrict__ out, int n) {
  for (int i = blockIdx.x * blockDim.x + threadIdx.x; i < n; i += gridDim.x * blockDim.x)
    out[i] = (__bf16)in[i];
}

// ---------- kernel 1: qkv = x @ Wqkv^T + b ; scatter q(scaled), k, v^T ----------

__global__ void qkv_gemm(const __bf16* __restrict__ xb, const __bf16* __restrict__ wb,
                         const float* __restrict__ bias,
                         __bf16* __restrict__ qb, __bf16* __restrict__ kbuf,
                         __bf16* __restrict__ vtb) {
  const int lane = threadIdx.x & 31;
  const int wv   = threadIdx.x >> 5;
  const int tile = blockIdx.x * 8 + wv;       // 512 * 36 = 18432 wave tiles
  const int NT   = QKV_OUT / 64;              // 36
  const int mt = tile / NT, nt = tile % NT;
  const int m0 = mt * 16, o0 = nt * 64;

  v8f acc[4] = {};
  for (int k0 = 0; k0 < EMB; k0 += 32) {
    v16bf a = load_a_frag(xb + (size_t)m0 * EMB + k0, EMB, lane);
    __builtin_prefetch(xb + (size_t)m0 * EMB + k0 + 32, 0, 3);
#pragma unroll
    for (int t = 0; t < 4; ++t) {
      v16bf b = load_b_frag(wb + (size_t)(o0 + 16 * t) * EMB + k0, EMB, lane);
      acc[t] = WMMA_BF16(a, b, acc[t]);
    }
  }

  const int hi = (lane >> 4) * 8, col = lane & 15;
#pragma unroll
  for (int t = 0; t < 4; ++t) {
    const int o   = o0 + 16 * t + col;
    const int tri = o / EMB, rem = o % EMB;
    const int h = rem >> 6, d = rem & 63;
    const float bv = bias[o];
#pragma unroll
    for (int r = 0; r < 8; ++r) {
      const int m  = m0 + hi + r;
      const int bq = m >> 11, n = m & (SEQ - 1);
      const size_t bh = (size_t)bq * HEADS + h;
      const float v = acc[t][r] + bv;
      if (tri == 0)      qb  [(bh * SEQ + n) * HD + d] = (__bf16)(v * 0.125f); // fold 1/sqrt(64)
      else if (tri == 1) kbuf[(bh * SEQ + n) * HD + d] = (__bf16)v;
      else               vtb [(bh * HD + d) * SEQ + n] = (__bf16)v;            // V transposed
    }
  }
}

// ---------- kernel 2: streaming (flash) attention, 16 query rows per wave ----------

__global__ void flash_attn(const __bf16* __restrict__ qb, const __bf16* __restrict__ kbuf,
                           const __bf16* __restrict__ vtb, __bf16* __restrict__ attnb) {
  __shared__ __align__(16) __bf16 Plds[8][16 * 32];   // per-wave 16x32 P transpose staging
  const int lane = threadIdx.x & 31;
  const int wv   = threadIdx.x >> 5;
  const int wid  = blockIdx.x * 8 + wv;               // 48 * 128 = 6144 waves
  const int bh = wid >> 7;
  const int n0 = (wid & 127) * 16;
  const int b = bh / HEADS, h = bh % HEADS;
  const size_t qoff = (size_t)bh * SEQ * HD;
  const int hi = (lane >> 4) * 8, col = lane & 15;

  const v16bf qa0 = load_a_frag(qb + qoff + (size_t)n0 * HD,      HD, lane); // d = 0..31
  const v16bf qa1 = load_a_frag(qb + qoff + (size_t)n0 * HD + 32, HD, lane); // d = 32..63

  v8f acc[4] = {};
  float m8[8], l8[8];
#pragma unroll
  for (int r = 0; r < 8; ++r) { m8[r] = -1e30f; l8[r] = 0.0f; }

  for (int mk = 0; mk < SEQ; mk += 32) {
    const __bf16* kp = kbuf + qoff + (size_t)mk * HD;
    v8f s0 = {}, s1 = {};
    s0 = WMMA_BF16(qa0, load_b_frag(kp,                HD, lane), s0);
    s0 = WMMA_BF16(qa1, load_b_frag(kp + 32,           HD, lane), s0);
    s1 = WMMA_BF16(qa0, load_b_frag(kp + 16 * HD,      HD, lane), s1);
    s1 = WMMA_BF16(qa1, load_b_frag(kp + 16 * HD + 32, HD, lane), s1);

    float alpha[8];
#pragma unroll
    for (int r = 0; r < 8; ++r) {
      // row-max over the 32 keys of this tile (butterfly within each 16-lane half)
      float t = fmaxf(s0[r], s1[r]);
      t = fmaxf(t, __shfl_xor(t, 1, 32));
      t = fmaxf(t, __shfl_xor(t, 2, 32));
      t = fmaxf(t, __shfl_xor(t, 4, 32));
      t = fmaxf(t, __shfl_xor(t, 8, 32));
      const float mn = fmaxf(m8[r], t);
      const float al = __expf(m8[r] - mn);
      const float p0 = __expf(s0[r] - mn);
      const float p1 = __expf(s1[r] - mn);
      float rs = p0 + p1;
      rs += __shfl_xor(rs, 1, 32);
      rs += __shfl_xor(rs, 2, 32);
      rs += __shfl_xor(rs, 4, 32);
      rs += __shfl_xor(rs, 8, 32);
      m8[r] = mn;
      l8[r] = l8[r] * al + rs;
      alpha[r] = al;
      Plds[wv][(hi + r) * 32 + col]      = (__bf16)p0;   // C-layout -> row-major LDS
      Plds[wv][(hi + r) * 32 + col + 16] = (__bf16)p1;
    }
#pragma unroll
    for (int t = 0; t < 4; ++t)
#pragma unroll
      for (int r = 0; r < 8; ++r) acc[t][r] *= alpha[r];

    // P (16x32) as A fragment from LDS; V^T supplies contiguous B fragments
    {
      const int row = lane & 15, kb = (lane >> 4) << 3;
      v8bf lo  = *(const v8bf*)&Plds[wv][row * 32 + kb];
      v8bf hiq = *(const v8bf*)&Plds[wv][row * 32 + kb + 16];
      const v16bf pa = cat8(lo, hiq);
      const __bf16* vp = vtb + (size_t)bh * HD * SEQ + mk;
#pragma unroll
      for (int t = 0; t < 4; ++t) {
        v16bf bfr = load_b_frag(vp + (size_t)(16 * t) * SEQ, SEQ, lane);
        acc[t] = WMMA_BF16(pa, bfr, acc[t]);
      }
    }
  }

#pragma unroll
  for (int r = 0; r < 8; ++r) l8[r] = 1.0f / l8[r];
#pragma unroll
  for (int t = 0; t < 4; ++t) {
    const int d = h * HD + 16 * t + col;
#pragma unroll
    for (int r = 0; r < 8; ++r) {
      const int n = n0 + hi + r;
      attnb[((size_t)b * SEQ + n) * EMB + d] = (__bf16)(acc[t][r] * l8[r]);
    }
  }
}

// ---------- kernel 3: out = attn @ Wproj^T + b (fp32 out) ----------

__global__ void proj_gemm(const __bf16* __restrict__ ab, const __bf16* __restrict__ wb,
                          const float* __restrict__ bias, float* __restrict__ out) {
  const int lane = threadIdx.x & 31;
  const int wv   = threadIdx.x >> 5;
  const int tile = blockIdx.x * 8 + wv;       // 512 * 12 = 6144
  const int NT   = EMB / 64;                  // 12
  const int mt = tile / NT, nt = tile % NT;
  const int m0 = mt * 16, o0 = nt * 64;

  v8f acc[4] = {};
  for (int k0 = 0; k0 < EMB; k0 += 32) {
    v16bf a = load_a_frag(ab + (size_t)m0 * EMB + k0, EMB, lane);
    __builtin_prefetch(ab + (size_t)m0 * EMB + k0 + 32, 0, 3);
#pragma unroll
    for (int t = 0; t < 4; ++t) {
      v16bf b = load_b_frag(wb + (size_t)(o0 + 16 * t) * EMB + k0, EMB, lane);
      acc[t] = WMMA_BF16(a, b, acc[t]);
    }
  }
  const int hi = (lane >> 4) * 8, col = lane & 15;
#pragma unroll
  for (int t = 0; t < 4; ++t) {
    const int o = o0 + 16 * t + col;
    const float bv = bias[o];
#pragma unroll
    for (int r = 0; r < 8; ++r) {
      const int m = m0 + hi + r;
      out[(size_t)m * EMB + o] = acc[t][r] + bv;
    }
  }
}

// ---------- host ----------

extern "C" void kernel_launch(void* const* d_in, const int* in_sizes, int n_in,
                              void* d_out, int out_size, void* d_ws, size_t ws_size,
                              hipStream_t stream) {
  const float* x      = (const float*)d_in[0];
  const float* w_qkv  = (const float*)d_in[1];
  const float* b_qkv  = (const float*)d_in[2];
  const float* w_proj = (const float*)d_in[3];
  const float* b_proj = (const float*)d_in[4];
  float* out = (float*)d_out;

  char* ws = (char*)d_ws;
  auto alloc = [&](size_t elems) -> __bf16* {
    __bf16* p = (__bf16*)ws;
    ws += ((elems * sizeof(__bf16) + 255) / 256) * 256;
    return p;
  };
  __bf16* xb     = alloc((size_t)M_ROWS * EMB);          // 8192x768
  __bf16* wqkvb  = alloc((size_t)QKV_OUT * EMB);         // 2304x768
  __bf16* wprojb = alloc((size_t)EMB * EMB);             // 768x768
  __bf16* qb     = alloc((size_t)BATCH * HEADS * SEQ * HD);
  __bf16* kb     = alloc((size_t)BATCH * HEADS * SEQ * HD);
  __bf16* vtb    = alloc((size_t)BATCH * HEADS * SEQ * HD);
  __bf16* ab     = alloc((size_t)M_ROWS * EMB);
  (void)ws_size; (void)in_sizes; (void)n_in; (void)out_size;

  f32_to_bf16<<<4096, 256, 0, stream>>>(x,      xb,     M_ROWS * EMB);
  f32_to_bf16<<<2048, 256, 0, stream>>>(w_qkv,  wqkvb,  QKV_OUT * EMB);
  f32_to_bf16<<<1024, 256, 0, stream>>>(w_proj, wprojb, EMB * EMB);

  qkv_gemm <<<2304, 256, 0, stream>>>(xb, wqkvb, b_qkv, qb, kb, vtb);
  flash_attn<<<768, 256, 0, stream>>>(qb, kb, vtb, ab);
  proj_gemm <<<768, 256, 0, stream>>>(ab, wprojb, b_proj, out);
}
